// DogeCDMoE_49228915146944
// MI455X (gfx1250) — matrix operand order repaired and checked by reading, not compiled, not run on hardware
//
#include <hip/hip_runtime.h>
#include <hip/hip_bf16.h>

#define HID    1024
#define INTER  4096
#define NTOK   4096   // B*S = 2*2048
#define NKEYS  64
#define TOPK   8

typedef __attribute__((ext_vector_type(16))) __bf16 v16bf;
typedef __attribute__((ext_vector_type(8)))  __bf16 v8bf;
typedef __attribute__((ext_vector_type(8)))  float  v8f;

// ---------- helpers ----------
__device__ __forceinline__ unsigned short f2bfu(float f) {
    unsigned u = __builtin_bit_cast(unsigned, f);
    u += 0x7FFFu + ((u >> 16) & 1u);           // round-to-nearest-even
    return (unsigned short)(u >> 16);
}
__device__ __forceinline__ float bf2f(unsigned short s) {
    unsigned u = ((unsigned)s) << 16;
    return __builtin_bit_cast(float, u);
}
__device__ __forceinline__ float bfe2f(__bf16 h) {
    return bf2f(__builtin_bit_cast(unsigned short, h));
}

// =====================================================================
// fp32 -> bf16 bulk convert (bandwidth-bound pre-pass; count % 4 == 0)
// =====================================================================
__global__ __launch_bounds__(256) void f32_to_bf16_kernel(const float* __restrict__ in,
                                                          unsigned short* __restrict__ out,
                                                          int count)
{
    int i = (blockIdx.x * 256 + threadIdx.x) * 4;
    if (i < count) {
        float4 v = *(const float4*)(in + i);
        ushort4 r;
        r.x = f2bfu(v.x); r.y = f2bfu(v.y); r.z = f2bfu(v.z); r.w = f2bfu(v.w);
        *(ushort4*)(out + i) = r;
    }
}

// =====================================================================
// All-bf16 WMMA NT GEMM:  C[M,N] = A[M,K] * B[N,K]^T
// A, B: bf16 row-major.  C: fp32 or bf16.
// Block = 128 threads (4 waves, 2x2), block tile 64x128, wave tile 32x64.
// Requires M%64==0, N%128==0, K%64==0 (true for all uses here).
// 16-bit A 16x32 / B 32x16 fragments: contiguous k = half*16 + s; f32 C 16x16 as in CDNA5 ISA 7.12.2.
// Inner loop (unroll 2): 24x global_load_b128 + 16x v_wmma_f32_16x16x32_bf16.
// =====================================================================
template <bool OUT_IS_BF16>
__global__ __launch_bounds__(128) void wmma_gemm_nt(const unsigned short* __restrict__ Au,
                                                    const unsigned short* __restrict__ Bu,
                                                    void* __restrict__ Cv,
                                                    int M, int N, int K)
{
    const int lane = threadIdx.x & 31;
    const int wave = threadIdx.x >> 5;
    const int half = lane >> 4;   // 0: lanes 0-15, 1: lanes 16-31
    const int l16  = lane & 15;
    const int m0 = blockIdx.y * 64  + (wave >> 1) * 32;
    const int n0 = blockIdx.x * 128 + (wave & 1) * 64;

    const __bf16* Ab = (const __bf16*)Au;
    const __bf16* Bb = (const __bf16*)Bu;

    v8f acc[2][4];
#pragma unroll
    for (int mi = 0; mi < 2; ++mi)
#pragma unroll
        for (int ni = 0; ni < 4; ++ni)
#pragma unroll
            for (int e = 0; e < 8; ++e) acc[mi][ni][e] = 0.f;

    // Per-lane base pointers (advance by 32 elements of K per step)
    const __bf16* aptr[2];
#pragma unroll
    for (int mi = 0; mi < 2; ++mi)
        aptr[mi] = Ab + (size_t)(m0 + mi * 16 + l16) * K + half * 16;
    const __bf16* bptr[4];
#pragma unroll
    for (int ni = 0; ni < 4; ++ni)
        bptr[ni] = Bb + (size_t)(n0 + ni * 16 + l16) * K + half * 16;

#pragma unroll 2
    for (int k0 = 0; k0 < K; k0 += 32) {
        // ---- A fragments: 16 contiguous K starting at kb = half*16
        v16bf a[2];
#pragma unroll
        for (int mi = 0; mi < 2; ++mi) {
            v8bf lo = *(const v8bf*)(aptr[mi] + k0);
            v8bf hi = *(const v8bf*)(aptr[mi] + k0 + 8);
            a[mi] = __builtin_shufflevector(lo, hi, 0, 1, 2, 3, 4, 5, 6, 7,
                                            8, 9, 10, 11, 12, 13, 14, 15);
        }
        // ---- B fragments: 16 contiguous K starting at klo = half*16
        v16bf b[4];
#pragma unroll
        for (int ni = 0; ni < 4; ++ni)
            b[ni] = *(const v16bf*)(bptr[ni] + k0);
        // ---- 8 WMMAs
#pragma unroll
        for (int mi = 0; mi < 2; ++mi)
#pragma unroll
            for (int ni = 0; ni < 4; ++ni)
                acc[mi][ni] = __builtin_amdgcn_wmma_f32_16x16x32_bf16(
                    false, a[mi], false, b[ni], (short)0, acc[mi][ni], false, false);
    }

    // ---- store: C/D layout — lane: n = l16, row = half*8 + j
#pragma unroll
    for (int mi = 0; mi < 2; ++mi) {
#pragma unroll
        for (int ni = 0; ni < 4; ++ni) {
            const int nn = n0 + ni * 16 + l16;
#pragma unroll
            for (int j = 0; j < 8; ++j) {
                const int m = m0 + mi * 16 + half * 8 + j;
                if constexpr (OUT_IS_BF16)
                    ((unsigned short*)Cv)[(size_t)m * N + nn] = f2bfu(acc[mi][ni][j]);
                else
                    ((float*)Cv)[(size_t)m * N + nn] = acc[mi][ni][j];
            }
        }
    }
}

// =====================================================================
// H = silu(G) * U   (bf16 in, bf16 out, in-place into G allowed)
// 8 elements per thread -> b128 loads/stores per lane.
// =====================================================================
__global__ __launch_bounds__(256) void silu_mul_kernel(const unsigned short* __restrict__ G,
                                                       const unsigned short* __restrict__ U,
                                                       unsigned short* __restrict__ H,
                                                       int count)
{
    int i = (blockIdx.x * 256 + threadIdx.x) * 8;
    if (i < count) {
        v8bf g8 = *(const v8bf*)((const __bf16*)G + i);
        v8bf u8 = *(const v8bf*)((const __bf16*)U + i);
        ushort4 r0, r1;
        float g, u;
        g = bfe2f(g8[0]); u = bfe2f(u8[0]); r0.x = f2bfu(g / (1.f + __expf(-g)) * u);
        g = bfe2f(g8[1]); u = bfe2f(u8[1]); r0.y = f2bfu(g / (1.f + __expf(-g)) * u);
        g = bfe2f(g8[2]); u = bfe2f(u8[2]); r0.z = f2bfu(g / (1.f + __expf(-g)) * u);
        g = bfe2f(g8[3]); u = bfe2f(u8[3]); r0.w = f2bfu(g / (1.f + __expf(-g)) * u);
        g = bfe2f(g8[4]); u = bfe2f(u8[4]); r1.x = f2bfu(g / (1.f + __expf(-g)) * u);
        g = bfe2f(g8[5]); u = bfe2f(u8[5]); r1.y = f2bfu(g / (1.f + __expf(-g)) * u);
        g = bfe2f(g8[6]); u = bfe2f(u8[6]); r1.z = f2bfu(g / (1.f + __expf(-g)) * u);
        g = bfe2f(g8[7]); u = bfe2f(u8[7]); r1.w = f2bfu(g / (1.f + __expf(-g)) * u);
        *(ushort4*)(H + i)     = r0;
        *(ushort4*)(H + i + 4) = r1;
    }
}

// =====================================================================
// Routing: per token, top-8 of x-list & y-list (64 each), top-8 of the
// 64 pairwise sums, softmax.  One wave32 per token, stable tie-break.
// =====================================================================
__device__ __forceinline__ void wave_argmax(float& bv, int& bi)
{
#pragma unroll
    for (int off = 16; off; off >>= 1) {
        float ov = __shfl_xor(bv, off, 32);
        int   oi = __shfl_xor(bi, off, 32);
        if (ov > bv || (ov == bv && oi < bi)) { bv = ov; bi = oi; }
    }
}

__global__ __launch_bounds__(128) void route_topk(const float* __restrict__ logits,
                                                  int* __restrict__ idx_out,
                                                  float* __restrict__ w_out,
                                                  int n)
{
    __shared__ float s_sx[4][8], s_sy[4][8];
    __shared__ int   s_ix[4][8], s_iy[4][8];
    const int w    = threadIdx.x >> 5;
    const int lane = threadIdx.x & 31;
    int t = blockIdx.x * 4 + w;
    if (t >= n) t = 0;   // grid is exact; defensive only (duplicate work is deterministic)

    const float NEG_INF = -__builtin_inff();
    const float* xr = logits + (size_t)t * 64;
    const float* yr = logits + (size_t)n * 64 + (size_t)t * 64;
    float xv0 = xr[lane], xv1 = xr[lane + 32];
    float yv0 = yr[lane], yv1 = yr[lane + 32];

    // ---- top-8 of x-list with original positions
#pragma unroll
    for (int r = 0; r < 8; ++r) {
        float bv; int bi;
        if (xv0 >= xv1) { bv = xv0; bi = lane; } else { bv = xv1; bi = lane + 32; }
        wave_argmax(bv, bi);
        if (lane == 0) { s_sx[w][r] = bv; s_ix[w][r] = bi; }
        if (bi == lane)      xv0 = NEG_INF;
        if (bi == lane + 32) xv1 = NEG_INF;
    }
    // ---- top-8 of y-list
#pragma unroll
    for (int r = 0; r < 8; ++r) {
        float bv; int bi;
        if (yv0 >= yv1) { bv = yv0; bi = lane; } else { bv = yv1; bi = lane + 32; }
        wave_argmax(bv, bi);
        if (lane == 0) { s_sy[w][r] = bv; s_iy[w][r] = bi; }
        if (bi == lane)      yv0 = NEG_INF;
        if (bi == lane + 32) yv1 = NEG_INF;
    }

    // ---- 64 pairwise sums, pair id p = ri*8+rj (matches reference flat order)
    float p0 = s_sx[w][lane >> 3]       + s_sy[w][lane & 7];
    float p1 = s_sx[w][(lane >> 3) + 4] + s_sy[w][lane & 7];
    const int pi0 = lane, pi1 = lane + 32;

    float ts[8]; int ex[8];
#pragma unroll
    for (int r = 0; r < 8; ++r) {
        float bv; int bi;
        if (p0 >= p1) { bv = p0; bi = pi0; } else { bv = p1; bi = pi1; }
        wave_argmax(bv, bi);
        ts[r] = bv;
        ex[r] = s_ix[w][bi >> 3] * 64 + s_iy[w][bi & 7];
        if (bi == pi0) p0 = NEG_INF;
        if (bi == pi1) p1 = NEG_INF;
    }

    // ---- softmax over the 8 (ts[0] is the max; extra renorm is a no-op)
    float mx = ts[0], e[8], sum = 0.f;
#pragma unroll
    for (int r = 0; r < 8; ++r) { e[r] = __expf(ts[r] - mx); sum += e[r]; }
    if (lane == 0) {
        const float inv = 1.f / sum;
#pragma unroll
        for (int r = 0; r < 8; ++r) {
            idx_out[t * 8 + r] = ex[r];
            w_out[t * 8 + r]   = e[r] * inv;
        }
    }
}

// =====================================================================
// Expert gather/combine + final add:
// out[t] = mlp[t] + sum_k silu(<de[e_k], x[t]>) * w_k * ue[e_k]
// One 256-thread block per token; 4 hidden dims per thread.
// =====================================================================
__global__ __launch_bounds__(256) void expert_combine(const float* __restrict__ x,
                                                      const float* __restrict__ de,
                                                      const float* __restrict__ ue,
                                                      const int* __restrict__ idx,
                                                      const float* __restrict__ wts,
                                                      const float* __restrict__ mlp,
                                                      float* __restrict__ out)
{
    const int t   = blockIdx.x;
    const int tid = threadIdx.x;
    __shared__ int   se[8];
    __shared__ float sw[8];
    __shared__ float red[8][8];
    __shared__ float ew[8];
    if (tid < 8) { se[tid] = idx[t * 8 + tid]; sw[tid] = wts[t * 8 + tid]; }
    __syncthreads();

    const float* xp = x + (size_t)t * HID;
    float xv[4];
#pragma unroll
    for (int j = 0; j < 4; ++j) xv[j] = xp[tid + 256 * j];

    float part[8];
#pragma unroll
    for (int k = 0; k < 8; ++k) {
        const float* dp = de + (size_t)se[k] * HID;
        float s = 0.f;
#pragma unroll
        for (int j = 0; j < 4; ++j) s += dp[tid + 256 * j] * xv[j];
        part[k] = s;
    }
#pragma unroll
    for (int k = 0; k < 8; ++k)
#pragma unroll
        for (int off = 16; off; off >>= 1) part[k] += __shfl_xor(part[k], off, 32);

    const int lane = tid & 31, wv = tid >> 5;
    if (lane == 0) {
#pragma unroll
        for (int k = 0; k < 8; ++k) red[wv][k] = part[k];
    }
    __syncthreads();
    if (tid < 8) {
        float s = 0.f;
#pragma unroll
        for (int w2 = 0; w2 < 8; ++w2) s += red[w2][tid];
        ew[tid] = s / (1.f + __expf(-s)) * sw[tid];
    }
    __syncthreads();

    float acc[4] = {0.f, 0.f, 0.f, 0.f};
#pragma unroll
    for (int k = 0; k < 8; ++k) {
        const float* up = ue + (size_t)se[k] * HID;
        const float g = ew[k];
#pragma unroll
        for (int j = 0; j < 4; ++j) acc[j] += g * up[tid + 256 * j];
    }
    const float* mp = mlp + (size_t)t * HID;
    float*       op = out + (size_t)t * HID;
#pragma unroll
    for (int j = 0; j < 4; ++j) op[tid + 256 * j] = mp[tid + 256 * j] + acc[j];
}

// =====================================================================
extern "C" void kernel_launch(void* const* d_in, const int* in_sizes, int n_in,
                              void* d_out, int out_size, void* d_ws, size_t ws_size,
                              hipStream_t stream)
{
    (void)in_sizes; (void)n_in; (void)out_size; (void)ws_size;
    const float* x  = (const float*)d_in[0];  // (n, HID)
    const float* wg = (const float*)d_in[1];  // (INTER, HID)
    const float* wu = (const float*)d_in[2];  // (INTER, HID)
    const float* wd = (const float*)d_in[3];  // (HID, INTER)
    const float* wr = (const float*)d_in[4];  // (2*NKEYS, HID)
    const float* de = (const float*)d_in[5];  // (NUM_EXPERTS, HID)
    const float* ue = (const float*)d_in[6];  // (NUM_EXPERTS, HID)

    float* out    = (float*)d_out;                          // (n, HID)
    float* logits = out + (size_t)NTOK * HID;               // (n, 128) == reshape(2,n,64) flat

    char* ws = (char*)d_ws;
    unsigned short* Gb  = (unsigned short*)(ws);                       // 32 MB (also H)
    unsigned short* Ub  = (unsigned short*)(ws + (size_t)33554432);    // 32 MB
    float*          mlp = (float*)(ws + (size_t)67108864);             // 16 MB
    unsigned short* Xb  = (unsigned short*)(ws + (size_t)83886080);    // 8 MB
    unsigned short* Wgb = (unsigned short*)(ws + (size_t)92274688);    // 8 MB
    unsigned short* Wub = (unsigned short*)(ws + (size_t)100663296);   // 8 MB
    unsigned short* Wdb = (unsigned short*)(ws + (size_t)109051904);   // 8 MB
    unsigned short* Wrb = (unsigned short*)(ws + (size_t)117440512);   // 256 KB
    int*            eix = (int*)  (ws + (size_t)117702656);            // 128 KB
    float*          ewt = (float*)(ws + (size_t)117833728);            // 128 KB

    // 0) one-time fp32 -> bf16 conversions (bandwidth-bound, ~6 us total)
    const int cX = NTOK * HID, cW = INTER * HID, cR = 2 * NKEYS * HID;
    f32_to_bf16_kernel<<<cX / 1024, 256, 0, stream>>>(x,  Xb,  cX);
    f32_to_bf16_kernel<<<cW / 1024, 256, 0, stream>>>(wg, Wgb, cW);
    f32_to_bf16_kernel<<<cW / 1024, 256, 0, stream>>>(wu, Wub, cW);
    f32_to_bf16_kernel<<<cW / 1024, 256, 0, stream>>>(wd, Wdb, cW);
    f32_to_bf16_kernel<<<cR / 1024, 256, 0, stream>>>(wr, Wrb, cR);

    const dim3 blk(128);

    // 1) router logits (fp32 out, straight into d_out tail): M=4096,N=128,K=1024
    wmma_gemm_nt<false><<<dim3(1, NTOK / 64), blk, 0, stream>>>(
        Xb, Wrb, (void*)logits, NTOK, 2 * NKEYS, HID);

    // 2) G = X*Wg^T, U = X*Wu^T  (bf16 out): M=4096,N=4096,K=1024
    wmma_gemm_nt<true><<<dim3(INTER / 128, NTOK / 64), blk, 0, stream>>>(
        Xb, Wgb, (void*)Gb, NTOK, INTER, HID);
    wmma_gemm_nt<true><<<dim3(INTER / 128, NTOK / 64), blk, 0, stream>>>(
        Xb, Wub, (void*)Ub, NTOK, INTER, HID);

    // 3) H = silu(G)*U (in-place into G buffer)
    const int cnt = NTOK * INTER;
    silu_mul_kernel<<<cnt / 2048, 256, 0, stream>>>(Gb, Ub, Gb, cnt);

    // 4) mlp = H*Wd^T : M=4096,N=1024,K=4096 (fp32 out)
    wmma_gemm_nt<false><<<dim3(HID / 128, NTOK / 64), blk, 0, stream>>>(
        Gb, Wdb, (void*)mlp, NTOK, HID, INTER);

    // 5) routing top-k (one wave per token)
    route_topk<<<NTOK / 4, 128, 0, stream>>>(logits, eix, ewt, NTOK);

    // 6) expert gather/combine + final add
    expert_combine<<<NTOK, 256, 0, stream>>>(x, de, ue, eix, ewt, mlp, out);
}
